// MemoryReader_29119878267571
// MI455X (gfx1250) — compile-verified, matrix operand order checked by reading, not run on hardware
//
#include <hip/hip_runtime.h>
#include <hip/hip_bf16.h>

typedef _Float16 h16;
typedef __attribute__((ext_vector_type(16))) _Float16 v16h;
typedef __attribute__((ext_vector_type(8)))  _Float16 v8h;
typedef __attribute__((ext_vector_type(8)))  float    v8f;

#define WMMA16(a, b, c) \
  __builtin_amdgcn_wmma_f32_16x16x32_f16(false, (a), false, (b), (short)0, (c), false, false)

// Problem constants (B, C_K, C_V, H, W) = (4, 64, 128, 64, 64)
#define NB   4
#define CK   64
#define CV   128
#define NPIX 4096           // 64*64
#define NCOL 16384          // NB*NPIX
#define BN_EPS 1e-5f

// Load a 16-half A/B fragment whose K-values are contiguous in memory starting at p.
// Layout (ISA 7.12.2, 16-bit 16x32): element i <-> K = (i&8)*2 + (half<<3) + (i&7),
// i.e. halves [half*8, half*8+7] then [16+half*8, 16+half*8+7]: two 16B vector loads.
__device__ __forceinline__ v16h load_frag_contig(const h16* p, int half) {
  const v8h lo = *reinterpret_cast<const v8h*>(p + (half << 3));
  const v8h hi = *reinterpret_cast<const v8h*>(p + 16 + (half << 3));
  v16h f;
#pragma unroll
  for (int i = 0; i < 8; ++i) { f[i] = lo[i]; f[8 + i] = hi[i]; }
  return f;
}

// Async DMA: 16 bytes global -> LDS, tracked by ASYNCcnt (CDNA5 §10 async ops).
__device__ __forceinline__ void async_b128(unsigned int lds_addr, const h16* g) {
  asm volatile("global_load_async_to_lds_b128 %0, %1, off"
               :: "v"(lds_addr), "v"((unsigned long long)(uintptr_t)g)
               : "memory");
}
__device__ __forceinline__ void wait_async0() {
  asm volatile("s_wait_asynccnt 0x0" ::: "memory");
}
__device__ __forceinline__ unsigned int lds_lo32(const h16* p) {
  return (unsigned int)(uintptr_t)p;   // flat->LDS mapping uses addr[31:0]
}

// ---------------- conversion kernels ----------------
// in[b][c][n] (f32) -> out[b][n][c] (f16), scaled
__global__ void convT_kernel(const float* __restrict__ in, h16* __restrict__ out,
                             float scale, int C, int total) {
  int tid = blockIdx.x * blockDim.x + threadIdx.x;
  if (tid >= total) return;
  int n = tid % NPIX;
  int c = (tid / NPIX) % C;
  int b = tid / (NPIX * C);
  out[((size_t)b * NPIX + n) * C + c] = (h16)(in[tid] * scale);
}

__global__ void conv_kernel(const float* __restrict__ in, h16* __restrict__ out, int total) {
  int tid = blockIdx.x * blockDim.x + threadIdx.x;
  if (tid < total) out[tid] = (h16)in[tid];
}

__global__ void zero_kernel(float* __restrict__ p, int n) {
  int tid = blockIdx.x * blockDim.x + threadIdx.x;
  if (tid < n) p[tid] = 0.0f;
}

// ---------------- flash attention (LDS double-buffered, async staging) ----------------
// KmT: [B][Nm][64] f16 (pre-scaled by 1/8), KqT: [B][Nq][64] f16,
// Vm: [B][128][Nm] f16 -> memh: [B][128][Nq] f16
// block = 128 threads = 4 waves; each wave owns 16 queries, all 128 value rows.
// Km/Vm m-step tiles are staged once per block into LDS via async copies.
__global__ void __launch_bounds__(128) attn_kernel(const h16* __restrict__ KmT,
                                                   const h16* __restrict__ KqT,
                                                   const h16* __restrict__ Vm,
                                                   h16* __restrict__ memh) {
  __shared__ __align__(16) h16 sKm[2][32 * CK];    // 32 m-rows x 64 ch   (4 KB each)
  __shared__ __align__(16) h16 sVm[2][CV * 32];    // 128 v-rows x 32 m   (8 KB each)

  const int tid  = threadIdx.x;
  const int lane = tid & 31;
  const int lq   = lane & 15;
  const int half = lane >> 4;
  const int wave = tid >> 5;
  const int b    = blockIdx.y;
  const int qbase = blockIdx.x * 64 + wave * 16;

  // Q fragments (B operand: K=c, N=q), loaded once from global
  const h16* qrow = KqT + ((size_t)b * NPIX + qbase + lq) * CK;
  const v16h q0 = load_frag_contig(qrow,      half);
  const v16h q1 = load_frag_contig(qrow + 32, half);

  const h16* kmb = KmT + (size_t)b * NPIX * CK;
  const h16* vmb = Vm  + (size_t)b * CV * NPIX;

  // Stage one m-step tile (Km: 256 chunks of 16B, Vm: 512 chunks) across 128 threads.
  auto issue = [&](int buf, int mb) {
#pragma unroll
    for (int k = 0; k < 2; ++k) {                 // Km tile is fully contiguous in global
      const int c = tid + k * 128;
      async_b128(lds_lo32(&sKm[buf][c * 8]), kmb + (size_t)mb * CK + c * 8);
    }
#pragma unroll
    for (int k = 0; k < 4; ++k) {                 // Vm rows: 64B each, contiguous in m
      const int c = tid + k * 128;
      const int v = c >> 2, off = (c & 3) * 8;
      async_b128(lds_lo32(&sVm[buf][c * 8]), vmb + (size_t)v * NPIX + mb + off);
    }
  };

  v8f acc[8];
#pragma unroll
  for (int t = 0; t < 8; ++t) acc[t] = v8f{0.f,0.f,0.f,0.f,0.f,0.f,0.f,0.f};
  float mrun = -3.0e38f, lrun = 0.0f;

  issue(0, 0);

#pragma unroll 1
  for (int mb = 0; mb < NPIX; mb += 32) {
    const int buf = (mb >> 5) & 1;
    wait_async0();        // my async copies into `buf` done
    __syncthreads();      // everyone's copies visible; prev readers of buf^1 done
    if (mb + 32 < NPIX) issue(buf ^ 1, mb + 32);   // overlap next DMA with compute

    // score tiles S0 (rows mb..mb+15) and S1 (rows mb+16..mb+31), cols = q
    const h16* k0 = &sKm[buf][lq * CK];
    const h16* k1 = &sKm[buf][(16 + lq) * CK];
    v16h a00 = load_frag_contig(k0,      half);
    v16h a01 = load_frag_contig(k0 + 32, half);
    v16h a10 = load_frag_contig(k1,      half);
    v16h a11 = load_frag_contig(k1 + 32, half);

    v8f s0 = v8f{0.f,0.f,0.f,0.f,0.f,0.f,0.f,0.f};
    v8f s1 = v8f{0.f,0.f,0.f,0.f,0.f,0.f,0.f,0.f};
    s0 = WMMA16(a00, q0, s0);
    s0 = WMMA16(a01, q1, s0);
    s1 = WMMA16(a10, q0, s1);
    s1 = WMMA16(a11, q1, s1);

    // online softmax over memory rows (column-wise): local max/sum + cross-half shuffle
    float mloc = -3.0e38f;
#pragma unroll
    for (int r = 0; r < 8; ++r) mloc = fmaxf(mloc, fmaxf(s0[r], s1[r]));
    mloc = fmaxf(mloc, __shfl_xor(mloc, 16, 32));
    const float mnew  = fmaxf(mrun, mloc);
    const float alpha = __expf(mrun - mnew);

    v16h p;
    float rs = 0.0f;
#pragma unroll
    for (int r = 0; r < 8; ++r) {
      const float e0 = __expf(s0[r] - mnew);
      const float e1 = __expf(s1[r] - mnew);
      p[r]     = (h16)e0;   // K = r + 8*half      (tile S0)
      p[r + 8] = (h16)e1;   // K = 16 + r + 8*half (tile S1)
      rs += e0 + e1;
    }
    rs += __shfl_xor(rs, 16, 32);
    lrun = lrun * alpha + rs;
    mrun = mnew;

#pragma unroll
    for (int t = 0; t < 8; ++t) acc[t] *= alpha;
#pragma unroll
    for (int t = 0; t < 8; ++t) {
      v16h av = load_frag_contig(&sVm[buf][(t * 16 + lq) * 32], half);  // A: M=v, K=m
      acc[t] = WMMA16(av, p, acc[t]);
    }
  }

  const float linv = 1.0f / lrun;
#pragma unroll
  for (int t = 0; t < 8; ++t) {
#pragma unroll
    for (int r = 0; r < 8; ++r) {
      const int v = t * 16 + r + 8 * half;           // C-layout row
      memh[((size_t)b * CV + v) * NPIX + qbase + lq] = (h16)(acc[t][r] * linv);
    }
  }
}

// ---------------- conv1x1 GEMM + BN batch-stat accumulation ----------------
// y[o][col] = sum_c W[o][c] * E[c][col],  E = concat(mem, V_Q) per batch.
// block = 256 threads = 8 waves; each wave owns 16 columns, all 128 outputs.
__global__ void __launch_bounds__(256) gemm2_kernel(const h16* __restrict__ Wh,
                                                    const h16* __restrict__ memh,
                                                    const h16* __restrict__ Vqh,
                                                    float* __restrict__ y,
                                                    float* __restrict__ sum,
                                                    float* __restrict__ sumsq) {
  const int lane = threadIdx.x & 31;
  const int lq   = lane & 15;
  const int half = lane >> 4;
  const int wave = threadIdx.x >> 5;
  const int col  = blockIdx.x * 128 + wave * 16 + lq;
  const int b    = col >> 12;       // col / 4096 (16-col strips never straddle a batch)
  const int n    = col & 4095;

  v8f acc[8];
#pragma unroll
  for (int t = 0; t < 8; ++t) acc[t] = v8f{0.f,0.f,0.f,0.f,0.f,0.f,0.f,0.f};

#pragma unroll
  for (int ks = 0; ks < 8; ++ks) {
    const int kbase = ks * 32;
    const h16* colptr = (kbase < CV)
        ? memh + ((size_t)b * CV + kbase) * NPIX + n
        : Vqh  + ((size_t)b * CV + (kbase - CV)) * NPIX + n;
    __builtin_prefetch(colptr + (size_t)32 * NPIX, 0, 1);
    v16h bf;
#pragma unroll
    for (int i = 0; i < 16; ++i) {
      const int c = ((i & 8) << 1) + (half << 3) + (i & 7);
      bf[i] = colptr[(size_t)c * NPIX];
    }
#pragma unroll
    for (int ot = 0; ot < 8; ++ot) {
      const h16* wrow = Wh + (size_t)(ot * 16 + lq) * 256 + kbase;  // A: M=o, K=c (contig)
      v16h aw = load_frag_contig(wrow, half);
      acc[ot] = WMMA16(aw, bf, acc[ot]);
    }
  }

#pragma unroll
  for (int ot = 0; ot < 8; ++ot) {
#pragma unroll
    for (int r = 0; r < 8; ++r) {
      const int o = ot * 16 + r + 8 * half;
      const float v = acc[ot][r];
      y[(size_t)o * NCOL + col] = v;
      float s1 = v, s2 = v * v;
#pragma unroll
      for (int off = 1; off < 16; off <<= 1) {
        s1 += __shfl_xor(s1, off, 32);
        s2 += __shfl_xor(s2, off, 32);
      }
      if (lq == 0) { atomicAdd(&sum[o], s1); atomicAdd(&sumsq[o], s2); }
    }
  }
}

// ---------------- BN finalize + apply ----------------
__global__ void bn_finalize_kernel(const float* __restrict__ sum, const float* __restrict__ sumsq,
                                   const float* __restrict__ gamma, const float* __restrict__ beta,
                                   float* __restrict__ scale, float* __restrict__ shift) {
  const int o = threadIdx.x;
  if (o >= CV) return;
  const float invN = 1.0f / (float)NCOL;
  const float mean = sum[o] * invN;
  const float var  = fmaxf(sumsq[o] * invN - mean * mean, 0.0f);
  const float sc   = gamma[o] * rsqrtf(var + BN_EPS);
  scale[o] = sc;
  shift[o] = beta[o] - mean * sc;
}

__global__ void bn_apply_kernel(const float* __restrict__ y,
                                const float* __restrict__ scale, const float* __restrict__ shift,
                                float* __restrict__ out, int total) {
  int tid = blockIdx.x * blockDim.x + threadIdx.x;
  if (tid >= total) return;
  const int n = tid & 4095;
  const int o = (tid >> 12) & 127;
  const int b = tid >> 19;
  const float v = y[(size_t)o * NCOL + b * NPIX + n] * scale[o] + shift[o];
  out[tid] = fmaxf(v, 0.0f);   // out layout: [B][128][H][W] == tid order
}

// ---------------- launch ----------------
extern "C" void kernel_launch(void* const* d_in, const int* in_sizes, int n_in,
                              void* d_out, int out_size, void* d_ws, size_t ws_size,
                              hipStream_t stream) {
  const float* K_M  = (const float*)d_in[0];
  const float* V_M  = (const float*)d_in[1];
  const float* K_Q  = (const float*)d_in[2];
  const float* V_Q  = (const float*)d_in[3];
  const float* Wc   = (const float*)d_in[4];
  const float* gam  = (const float*)d_in[5];
  const float* bet  = (const float*)d_in[6];
  float* out = (float*)d_out;

  char* ws = (char*)d_ws;
  const size_t OFF_KMT = 0;                              // 2 MB
  const size_t OFF_KQT = (size_t)2  << 20;               // 2 MB
  const size_t OFF_VM  = (size_t)4  << 20;               // 4 MB
  const size_t OFF_VQ  = (size_t)8  << 20;               // 4 MB
  const size_t OFF_W   = (size_t)12 << 20;               // 64 KB
  const size_t OFF_MEM = OFF_W  + ((size_t)1 << 16);     // 4 MB
  const size_t OFF_Y   = OFF_MEM + ((size_t)4 << 20);    // 8 MB
  const size_t OFF_SUM = OFF_Y  + ((size_t)8 << 20);
  const size_t OFF_SQ  = OFF_SUM + 512;
  const size_t OFF_SC  = OFF_SQ + 512;
  const size_t OFF_SH  = OFF_SC + 512;

  h16*   KmT   = (h16*)(ws + OFF_KMT);
  h16*   KqT   = (h16*)(ws + OFF_KQT);
  h16*   Vmh   = (h16*)(ws + OFF_VM);
  h16*   Vqh   = (h16*)(ws + OFF_VQ);
  h16*   Wh    = (h16*)(ws + OFF_W);
  h16*   memh  = (h16*)(ws + OFF_MEM);
  float* yf    = (float*)(ws + OFF_Y);
  float* sums  = (float*)(ws + OFF_SUM);
  float* sqs   = (float*)(ws + OFF_SQ);
  float* scl   = (float*)(ws + OFF_SC);
  float* shf   = (float*)(ws + OFF_SH);

  const int nK = NB * CK * NPIX;   // 1,048,576
  const int nV = NB * CV * NPIX;   // 2,097,152
  const int nW = CV * 256;         // 32,768

  // f32 -> f16 staging (K transposed to [n][c]; 1/sqrt(64) folded into K_M)
  convT_kernel<<<(nK + 255) / 256, 256, 0, stream>>>(K_M, KmT, 0.125f, CK, nK);
  convT_kernel<<<(nK + 255) / 256, 256, 0, stream>>>(K_Q, KqT, 1.0f,   CK, nK);
  conv_kernel <<<(nV + 255) / 256, 256, 0, stream>>>(V_M, Vmh, nV);
  conv_kernel <<<(nV + 255) / 256, 256, 0, stream>>>(V_Q, Vqh, nV);
  conv_kernel <<<(nW + 255) / 256, 256, 0, stream>>>(Wc,  Wh,  nW);
  zero_kernel <<<1, 256, 0, stream>>>(sums, 256);        // sums + sumsq contiguous

  // fused attention (flash style, WMMA f16 -> f32, async LDS staging)
  attn_kernel<<<dim3(NPIX / 64, NB), 128, 0, stream>>>(KmT, KqT, Vmh, memh);

  // conv1x1 GEMM + batch-stat accumulation
  gemm2_kernel<<<NCOL / 128, 256, 0, stream>>>(Wh, memh, Vqh, yf, sums, sqs);

  // BN finalize + apply + ReLU
  bn_finalize_kernel<<<1, 128, 0, stream>>>(sums, sqs, gam, bet, scl, shf);
  bn_apply_kernel<<<(out_size + 255) / 256, 256, 0, stream>>>(yf, scl, shf, out, out_size);
}
